// DeepSpeedBlockBaselineLayer_15032385536357
// MI455X (gfx1250) — matrix operand order repaired and checked by reading, not compiled, not run on hardware
//
#include <hip/hip_runtime.h>

typedef __attribute__((ext_vector_type(16))) __bf16 v16bf;
typedef __attribute__((ext_vector_type(8)))  float  v8f;
typedef __attribute__((ext_vector_type(4)))  unsigned int v4u;
typedef __attribute__((ext_vector_type(8)))  int v8i;
typedef __attribute__((ext_vector_type(4)))  int v4i;
typedef unsigned int uint32;

#define S_    1024
#define B_    2
#define T_    2048
#define HID_  2048
#define NH_   16
#define KVH_  4
#define HDIM_ 128
#define QKV_  3072
#define GROUP_ 768
#define FFN_  4096
#define E_    8
#define CAP_  512

#ifndef USE_TDM
#define USE_TDM 1
#endif
#if USE_TDM && __has_builtin(__builtin_amdgcn_tensor_load_to_lds)
#define HAVE_TDM 1
#else
#define HAVE_TDM 0
#endif

// ---------------------------------------------------------------------------
// WMMA helpers (CDNA5 wave32, 16x16x32 bf16 -> f32)
// ---------------------------------------------------------------------------
__device__ __forceinline__ v8f wmma_bf16(v16bf a, v16bf b, v8f c) {
  return __builtin_amdgcn_wmma_f32_16x16x32_bf16(false, a, false, b, (short)0, c,
                                                 false, false);
}

// A-matrix 16x32 bf16 tile, row-major, leading dim `ld`.
// Lane's 16 elements = one contiguous 16-element K run: [16*half, +16)
//  ->  exactly 2x b128 loads.
__device__ __forceinline__ v16bf lds_load_A16x32(const __bf16* p, int ld) {
  int lane = threadIdx.x & 31;
  int lh = lane >> 4;
  int m  = lane & 15;
  const __bf16* q = p + m * ld + 16 * lh;
  v16bf a;
  ((uint4*)&a)[0] = *(const uint4*)(q);
  ((uint4*)&a)[1] = *(const uint4*)(q + 8);
  return a;
}

// B-matrix 32x16 bf16 tile from LDS stored transposed as [N=16][K=32] (ld=32).
// Lane's 16 elements = one contiguous 16-element K run at K = 16*khalf.
__device__ __forceinline__ v16bf lds_load_Bt32x16(const __bf16* p, int ld) {
  int lane = threadIdx.x & 31;
  int kh = lane >> 4;
  int n  = lane & 15;
  const __bf16* q = p + n * ld + kh * 16;
  v16bf b;
  ((uint4*)&b)[0] = *(const uint4*)(q);
  ((uint4*)&b)[1] = *(const uint4*)(q + 8);
  return b;
}

#if HAVE_TDM
// Tensor Data Mover: DMA a [rows=128] x [cols=32] bf16 tile (row stride = lda
// elements) from global into LDS at byte offset lds_off, packed row-major.
// D# built per CDNA5 ISA ch.8: group0 {count=1, lds_addr, global_addr, type=2},
// group1 {data_size=2B, huge tensor dims (no OOB clamp), tile 32x128, stride0}.
__device__ __forceinline__ void tdm_load_a_tile(unsigned lds_off,
                                                const __bf16* gsrc,
                                                int lda_elems) {
  unsigned long long ga = (unsigned long long)(size_t)gsrc;
  v4u g0;
  g0[0] = 1u;                                   // count=1 (valid user D#)
  g0[1] = lds_off;                              // lds_addr (bytes)
  g0[2] = (unsigned)ga;                         // global_addr[31:0]
  g0[3] = (unsigned)((ga >> 32) & 0x01FFFFFFull) | (2u << 30);  // [56:32]|type=2
  v8i g1;
  g1[0] = (int)(1u << 16);                      // data_size=1 -> 2 bytes
  g1[1] = (int)0xFFFF0000u;                     // tensor_dim0 lo16 in [31:16]
  g1[2] = (int)0xFFFF7FFFu;                     // tensor_dim0 hi | tensor_dim1 lo
  g1[3] = (int)((32u << 16) | 0x7FFFu);         // tensor_dim1 hi | tile_dim0=32
  g1[4] = 128;                                  // tile_dim1=128, tile_dim2=0
  g1[5] = lda_elems;                            // tensor_dim0_stride[31:0]
  g1[6] = 0;
  g1[7] = 0;
  v4i z4 = {0, 0, 0, 0};
#if defined(__clang_major__) && (__clang_major__ >= 23)
  v8i z8 = {0, 0, 0, 0, 0, 0, 0, 0};
  __builtin_amdgcn_tensor_load_to_lds(g0, g1, z4, z4, z8, 0);
#else
  __builtin_amdgcn_tensor_load_to_lds(g0, g1, z4, z4, 0);
#endif
}
#endif

// ---------------------------------------------------------------------------
// Generic bf16 WMMA GEMM: C[M,N] = A[M,K](bf16) * B (fp32, converted in-flight)
// BLAYOUT 0: B is [N,K] row-major; BLAYOUT 1: B is [K,N] row-major.
// EPI 0: store bf16 | 1: fp32 residual add | 2: exact GELU -> bf16 | 3: fp32
// 128x128 block tile, BK=32, double-buffered LDS; A tile via TDM (wave 0
// issues, s_wait_tensorcnt before the consuming barrier), B via threads.
// ---------------------------------------------------------------------------
template <int BLAYOUT, int EPI>
__global__ __launch_bounds__(256) void gemm_bf16_kernel(
    const __bf16* __restrict__ A, int lda, long strideA,
    const float* __restrict__ B, int ldb, long strideB,
    void* __restrict__ Cout, int ldc, long strideC,
    const float* __restrict__ Res, int K) {
  __shared__ __bf16 shA[2][128 * 32];
  __shared__ __bf16 shB[2][128 * 32];  // stored as [N=128][K=32]

  int bn = blockIdx.x * 128;
  int bm = blockIdx.y * 128;
  long e = blockIdx.z;
  A += e * strideA;
  B += e * strideB;

  int tid  = threadIdx.x;
  int wave = tid >> 5;
  int wm   = wave >> 1;  // 0..3
  int wn   = wave & 1;   // 0..1

  auto stageA = [&](int buf, int k0) {
#if HAVE_TDM
    if (wave == 0) {
      tdm_load_a_tile((unsigned)(size_t)&shA[buf][0],
                      A + (long)bm * lda + k0, lda);
    }
#else
    int r = tid >> 1, h2 = tid & 1;
    const uint4* src = (const uint4*)(A + (long)(bm + r) * lda + k0 + h2 * 16);
    uint4* dst = (uint4*)(&shA[buf][r * 32 + h2 * 16]);
    dst[0] = src[0];
    dst[1] = src[1];
#endif
  };
  auto stageB = [&](int buf, int k0) {
    if (BLAYOUT == 0) {
      int n = tid >> 1, h2 = tid & 1;
      const float* src = B + (long)(bn + n) * ldb + k0 + h2 * 16;
      __bf16* dst = &shB[buf][n * 32 + h2 * 16];
      __builtin_prefetch(src + 32, 0, 0);  // next-next k-tile
#pragma unroll
      for (int i = 0; i < 16; ++i) dst[i] = (__bf16)src[i];
    } else {
      int n = tid & 127, kh = tid >> 7;
      __builtin_prefetch(&B[(long)(k0 + 32 + kh * 16) * ldb + bn + n], 0, 0);
#pragma unroll
      for (int kk = 0; kk < 16; ++kk) {
        int k = kh * 16 + kk;
        shB[buf][n * 32 + k] = (__bf16)B[(long)(k0 + k) * ldb + bn + n];
      }
    }
  };

  v8f zero8 = {0.f, 0.f, 0.f, 0.f, 0.f, 0.f, 0.f, 0.f};
  v8f acc[2][4];
#pragma unroll
  for (int i = 0; i < 2; ++i)
#pragma unroll
    for (int j = 0; j < 4; ++j) acc[i][j] = zero8;

  int niter = K >> 5;
  stageA(0, 0);
  stageB(0, 0);
  for (int it = 0; it < niter; ++it) {
    int cur = it & 1;
#if HAVE_TDM
    if (wave == 0) __builtin_amdgcn_s_wait_tensorcnt(0);
#endif
    __syncthreads();
    if (it + 1 < niter) {  // stage next tile into the other buffer
      stageA(cur ^ 1, (it + 1) << 5);
      stageB(cur ^ 1, (it + 1) << 5);
    }
    v16bf a0 = lds_load_A16x32(&shA[cur][(wm * 32 + 0) * 32], 32);
    v16bf a1 = lds_load_A16x32(&shA[cur][(wm * 32 + 16) * 32], 32);
#pragma unroll
    for (int j = 0; j < 4; ++j) {
      v16bf b = lds_load_Bt32x16(&shB[cur][(wn * 64 + j * 16) * 32], 32);
      acc[0][j] = wmma_bf16(a0, b, acc[0][j]);
      acc[1][j] = wmma_bf16(a1, b, acc[1][j]);
    }
  }

  // epilogue: C layout — lane&15 = column, (lane>>4)*8 + vgpr = row
  int lane = tid & 31;
  int col  = lane & 15;
  int rowb = (lane >> 4) * 8;
#pragma unroll
  for (int i = 0; i < 2; ++i)
#pragma unroll
    for (int j = 0; j < 4; ++j)
#pragma unroll
      for (int v = 0; v < 8; ++v) {
        long gr = bm + wm * 32 + i * 16 + rowb + v;
        long gc = bn + wn * 64 + j * 16 + col;
        float x = acc[i][j][v];
        long idx = e * strideC + gr * (long)ldc + gc;
        if (EPI == 0) {
          ((__bf16*)Cout)[idx] = (__bf16)x;
        } else if (EPI == 1) {
          ((float*)Cout)[idx] = Res[gr * (long)ldc + gc] + x;
        } else if (EPI == 2) {
          float g = 0.5f * x * (1.0f + erff(x * 0.70710678118654752f));
          ((__bf16*)Cout)[idx] = (__bf16)g;
        } else {
          ((float*)Cout)[idx] = x;
        }
      }
}

// ---------------------------------------------------------------------------
// LayerNorm: fp32 in -> bf16 out. One block per token.
// ---------------------------------------------------------------------------
__global__ __launch_bounds__(256) void ln_kernel(const float* __restrict__ x,
                                                 const float* __restrict__ w,
                                                 const float* __restrict__ bia,
                                                 __bf16* __restrict__ y) {
  __shared__ float rs[256], rs2[256];
  long t = blockIdx.x;
  const float* xr = x + t * HID_;
  float s = 0.f, s2 = 0.f;
  for (int h = threadIdx.x; h < HID_; h += 256) {
    float v = xr[h];
    s += v;
    s2 += v * v;
  }
  rs[threadIdx.x] = s;
  rs2[threadIdx.x] = s2;
  __syncthreads();
  for (int st = 128; st > 0; st >>= 1) {
    if (threadIdx.x < st) {
      rs[threadIdx.x] += rs[threadIdx.x + st];
      rs2[threadIdx.x] += rs2[threadIdx.x + st];
    }
    __syncthreads();
  }
  float mu = rs[0] * (1.0f / HID_);
  float var = rs2[0] * (1.0f / HID_) - mu * mu;
  float rstd = rsqrtf(var + 1e-5f);
  for (int h = threadIdx.x; h < HID_; h += 256)
    y[t * HID_ + h] = (__bf16)((xr[h] - mu) * rstd * w[h] + bia[h]);
}

// ---------------------------------------------------------------------------
// Flash-style causal attention: one wave per (qtile16, head, batch).
// Q*K^T and P*V via bf16 WMMA; online softmax; P transposed via 1KB LDS.
// ---------------------------------------------------------------------------
__global__ __launch_bounds__(32) void attn_kernel(const __bf16* __restrict__ qkv,
                                                  __bf16* __restrict__ attn_out) {
  __shared__ __bf16 sp[16 * 32];
  int q0 = blockIdx.x * 16;
  int h  = blockIdx.y;
  int b  = blockIdx.z;
  int kvh = h >> 2;  // Q_PER_KV = 4
  long qoff = (long)kvh * GROUP_ + (long)(h & 3) * HDIM_;
  long koff = (long)kvh * GROUP_ + 512;
  long voff = (long)kvh * GROUP_ + 640;

  int lane = threadIdx.x;
  int lh = lane >> 4;
  int mr = lane & 15;

  // Q tile 16x128 in A layout (4 chunks of K=32): 2x b128 per chunk
  v16bf qa[4];
  const __bf16* qbase = qkv + ((long)(q0 + mr) * B_ + b) * QKV_ + qoff + 16 * lh;
#pragma unroll
  for (int c = 0; c < 4; ++c) {
    ((uint4*)&qa[c])[0] = *(const uint4*)(qbase + c * 32);
    ((uint4*)&qa[c])[1] = *(const uint4*)(qbase + c * 32 + 8);
  }

  v8f zero8 = {0.f, 0.f, 0.f, 0.f, 0.f, 0.f, 0.f, 0.f};
  v8f o[8];
#pragma unroll
  for (int j = 0; j < 8; ++j) o[j] = zero8;
  float mrow[8], lrow[8];
#pragma unroll
  for (int v = 0; v < 8; ++v) {
    mrow[v] = -1e30f;
    lrow[v] = 0.f;
  }

  const float scale = 0.0883883476483184f;  // 1/sqrt(128)
  int kend = q0 + 16;
  for (int k0 = 0; k0 < kend; k0 += 32) {
    // scores: two 16x16 C tiles over key columns [k0, k0+32)
    v8f sc[2];
    sc[0] = zero8;
    sc[1] = zero8;
#pragma unroll
    for (int t2 = 0; t2 < 2; ++t2) {
      const __bf16* kb_base =
          qkv + ((long)(k0 + t2 * 16 + mr) * B_ + b) * QKV_ + koff + lh * 16;
#pragma unroll
      for (int c = 0; c < 4; ++c) {
        v16bf kb;  // B[d, n] = K[n, d]: one contiguous 16-elem d run per lane
        ((uint4*)&kb)[0] = *(const uint4*)(kb_base + c * 32);
        ((uint4*)&kb)[1] = *(const uint4*)(kb_base + c * 32 + 8);
        sc[t2] = wmma_bf16(qa[c], kb, sc[t2]);
      }
    }
    // online softmax update; rows live in vgpr index, spread over 16 lanes
#pragma unroll
    for (int v = 0; v < 8; ++v) {
      int grow = q0 + lh * 8 + v;
      float x0 = sc[0][v] * scale;
      float x1 = sc[1][v] * scale;
      if (k0 + mr > grow) x0 = -1e30f;
      if (k0 + 16 + mr > grow) x1 = -1e30f;
      float mx = fmaxf(x0, x1);
#pragma unroll
      for (int off2 = 1; off2 < 16; off2 <<= 1)
        mx = fmaxf(mx, __shfl_xor(mx, off2, 32));
      float mnew = fmaxf(mrow[v], mx);
      float p0 = __expf(x0 - mnew);
      float p1 = __expf(x1 - mnew);
      float resc = __expf(mrow[v] - mnew);
      float rs = p0 + p1;
#pragma unroll
      for (int off2 = 1; off2 < 16; off2 <<= 1) rs += __shfl_xor(rs, off2, 32);
      lrow[v] = lrow[v] * resc + rs;
      mrow[v] = mnew;
      sp[(lh * 8 + v) * 32 + mr] = (__bf16)p0;
      sp[(lh * 8 + v) * 32 + 16 + mr] = (__bf16)p1;
#pragma unroll
      for (int j = 0; j < 8; ++j) o[j][v] = o[j][v] * resc;
    }
    __syncthreads();
    v16bf pa = lds_load_A16x32(sp, 32);  // P as 16x32 A tile
#pragma unroll
    for (int j = 0; j < 8; ++j) {
      v16bf vb;  // B[n, d]: lane column = d, pairs over key token n
#pragma unroll
      for (int r = 0; r < 8; ++r) {
        int n0 = lh * 16 + 2 * r;
        vb[2 * r]     = qkv[((long)(k0 + n0) * B_ + b) * QKV_ + voff + j * 16 + mr];
        vb[2 * r + 1] = qkv[((long)(k0 + n0 + 1) * B_ + b) * QKV_ + voff + j * 16 + mr];
      }
      o[j] = wmma_bf16(pa, vb, o[j]);
    }
    __syncthreads();
  }
#pragma unroll
  for (int v = 0; v < 8; ++v) {
    float inv = 1.0f / lrow[v];
#pragma unroll
    for (int j = 0; j < 8; ++j)
      attn_out[((long)(q0 + lh * 8 + v) * B_ + b) * (NH_ * HDIM_) +
               (long)h * HDIM_ + j * 16 + mr] = (__bf16)(o[j][v] * inv);
  }
}

// ---------------------------------------------------------------------------
// Gate logits: one block (64 thr) per token; logits[t,e] = x[t,:] . gate_w[:,e]
// ---------------------------------------------------------------------------
__global__ __launch_bounds__(64) void gate_kernel(const __bf16* __restrict__ x,
                                                  const float* __restrict__ gw,
                                                  float* __restrict__ logits) {
  __shared__ float part[64];
  long t = blockIdx.x;
  int e2 = threadIdx.x & 7;
  int c  = threadIdx.x >> 3;
  float s = 0.f;
  const __bf16* xr = x + t * HID_;
  for (int hh = c * 256; hh < c * 256 + 256; ++hh)
    s += (float)xr[hh] * gw[hh * E_ + e2];
  part[threadIdx.x] = s;
  __syncthreads();
  if (threadIdx.x < E_) {
    float tot = 0.f;
    for (int i = 0; i < 8; ++i) tot += part[i * 8 + threadIdx.x];
    logits[t * E_ + threadIdx.x] = tot;
  }
}

// ---------------------------------------------------------------------------
// Top-2 routing with capacity (matches reference cumsum semantics).
// ---------------------------------------------------------------------------
__global__ __launch_bounds__(256) void route_kernel(
    const float* __restrict__ logits, int* idx1, int* idx2, float* g1, float* g2,
    int* pos1, int* pos2, int* slot_token) {
  int tid = threadIdx.x;
  for (int i = tid; i < E_ * CAP_; i += 256) slot_token[i] = -1;
  for (int t = tid; t < T_; t += 256) {
    float l[E_];
    float mx = -1e30f;
    for (int e2 = 0; e2 < E_; ++e2) {
      l[e2] = logits[t * E_ + e2];
      mx = fmaxf(mx, l[e2]);
    }
    float se = 0.f;
    for (int e2 = 0; e2 < E_; ++e2) se += __expf(l[e2] - mx);
    int i1 = 0;
    float b1 = l[0];
    for (int e2 = 1; e2 < E_; ++e2)
      if (l[e2] > b1) { b1 = l[e2]; i1 = e2; }
    int i2 = (i1 == 0) ? 1 : 0;
    float b2 = l[i2];
    for (int e2 = 0; e2 < E_; ++e2)
      if (e2 != i1 && l[e2] > b2) { b2 = l[e2]; i2 = e2; }
    idx1[t] = i1;
    idx2[t] = i2;
    g1[t] = __expf(l[i1] - mx) / se;
    g2[t] = __expf(l[i2] - mx) / se;
  }
  __syncthreads();
  if (tid < E_) {
    int e2 = tid, c = 0;
    for (int t = 0; t < T_; ++t)
      if (idx1[t] == e2) {
        pos1[t] = c;
        if (c < CAP_) slot_token[e2 * CAP_ + c] = t;
        ++c;
      }
    int base = c, c2 = 0;  // loc2 offset = uncapped mask1 count (reference)
    for (int t = 0; t < T_; ++t)
      if (idx2[t] == e2) {
        int p = base + c2;
        pos2[t] = p;
        if (p < CAP_) slot_token[e2 * CAP_ + p] = t;
        ++c2;
      }
  }
  __syncthreads();
  for (int t = tid; t < T_; t += 256) {
    bool k1 = pos1[t] < CAP_;
    bool k2 = pos2[t] < CAP_;
    float a = k1 ? g1[t] : 0.f;
    float bb = k2 ? g2[t] : 0.f;
    float d = fmaxf(a + bb, 1.1920929e-07f);
    g1[t] = a / d;
    g2[t] = bb / d;
  }
}

// ---------------------------------------------------------------------------
// Dispatch: gather ln2 rows into expert slots (zero-fill empty slots).
// ---------------------------------------------------------------------------
__global__ __launch_bounds__(256) void dispatch_kernel(
    const __bf16* __restrict__ ln2, const int* __restrict__ slot_token,
    __bf16* __restrict__ xe) {
  long slot = blockIdx.x;
  int t = slot_token[slot];
  uint32* dst = (uint32*)(xe + slot * HID_);
  if (t < 0) {
    for (int i = threadIdx.x; i < HID_ / 2; i += 256) dst[i] = 0u;
  } else {
    const uint32* src = (const uint32*)(ln2 + (long)t * HID_);
    for (int i = threadIdx.x; i < HID_ / 2; i += 256) dst[i] = src[i];
  }
}

// ---------------------------------------------------------------------------
// Combine: out = h1 + g1*eo[e1,p1] + g2*eo[e2,p2]
// ---------------------------------------------------------------------------
__global__ __launch_bounds__(256) void combine_kernel(
    const float* __restrict__ h1, const float* __restrict__ eo,
    const int* __restrict__ idx1, const int* __restrict__ idx2,
    const int* __restrict__ pos1, const int* __restrict__ pos2,
    const float* __restrict__ g1, const float* __restrict__ g2,
    float* __restrict__ out) {
  long t = blockIdx.x;
  int p1 = pos1[t], p2 = pos2[t];
  bool k1 = p1 < CAP_, k2 = p2 < CAP_;
  float a = k1 ? g1[t] : 0.f;
  float bb = k2 ? g2[t] : 0.f;
  long o1 = k1 ? ((long)idx1[t] * CAP_ + p1) * HID_ : 0;
  long o2 = k2 ? ((long)idx2[t] * CAP_ + p2) * HID_ : 0;
  for (int hh = threadIdx.x; hh < HID_; hh += 256) {
    float v = h1[t * HID_ + hh];
    if (k1) v += a * eo[o1 + hh];
    if (k2) v += bb * eo[o2 + hh];
    out[t * HID_ + hh] = v;
  }
}

// ---------------------------------------------------------------------------
extern "C" void kernel_launch(void* const* d_in, const int* in_sizes, int n_in,
                              void* d_out, int out_size, void* d_ws,
                              size_t ws_size, hipStream_t stream) {
  (void)in_sizes; (void)n_in; (void)out_size; (void)ws_size;
  const float* hidden = (const float*)d_in[0];
  const float* ln1_w  = (const float*)d_in[1];
  const float* ln1_b  = (const float*)d_in[2];
  const float* qkv_w  = (const float*)d_in[3];
  const float* proj_w = (const float*)d_in[4];
  const float* ln2_w  = (const float*)d_in[5];
  const float* ln2_b  = (const float*)d_in[6];
  const float* gate_w = (const float*)d_in[7];
  const float* w1     = (const float*)d_in[8];
  const float* w2     = (const float*)d_in[9];
  float* out = (float*)d_out;

  char* ws = (char*)d_ws;
  size_t off = 0;
  auto alloc = [&](size_t bytes) -> void* {
    void* p = ws + off;
    off += (bytes + 255) & ~(size_t)255;
    return p;
  };
  __bf16* ln1_bf   = (__bf16*)alloc((size_t)T_ * HID_ * 2);
  __bf16* qkv_bf   = (__bf16*)alloc((size_t)T_ * QKV_ * 2);
  __bf16* aout_bf  = (__bf16*)alloc((size_t)T_ * HID_ * 2);
  float*  h1       = (float*)alloc((size_t)T_ * HID_ * 4);
  __bf16* ln2_bf   = (__bf16*)alloc((size_t)T_ * HID_ * 2);
  float*  logits   = (float*)alloc((size_t)T_ * E_ * 4);
  int*    idx1     = (int*)alloc((size_t)T_ * 4);
  int*    idx2     = (int*)alloc((size_t)T_ * 4);
  int*    pos1     = (int*)alloc((size_t)T_ * 4);
  int*    pos2     = (int*)alloc((size_t)T_ * 4);
  float*  g1       = (float*)alloc((size_t)T_ * 4);
  float*  g2       = (float*)alloc((size_t)T_ * 4);
  int*    slot_tok = (int*)alloc((size_t)E_ * CAP_ * 4);
  __bf16* xe_bf    = (__bf16*)alloc((size_t)E_ * CAP_ * HID_ * 2);
  __bf16* he_bf    = (__bf16*)alloc((size_t)E_ * CAP_ * FFN_ * 2);
  float*  eo       = (float*)alloc((size_t)E_ * CAP_ * HID_ * 4);

  // 1. LN1
  ln_kernel<<<T_, 256, 0, stream>>>(hidden, ln1_w, ln1_b, ln1_bf);

  // 2. QKV = ln1 @ qkv_w^T  (B layout [N,K]) -> bf16
  gemm_bf16_kernel<0, 0><<<dim3(QKV_ / 128, T_ / 128, 1), 256, 0, stream>>>(
      ln1_bf, HID_, 0, qkv_w, HID_, 0, qkv_bf, QKV_, 0, nullptr, HID_);

  // 3. Attention (flash, causal, GQA repeat folded by kv-head indexing)
  attn_kernel<<<dim3(S_ / 16, NH_, B_), 32, 0, stream>>>(qkv_bf, aout_bf);

  // 4. h1 = hidden + attn_out @ proj_w^T  (B layout [N,K], fp32 residual)
  gemm_bf16_kernel<0, 1><<<dim3(HID_ / 128, T_ / 128, 1), 256, 0, stream>>>(
      aout_bf, NH_ * HDIM_, 0, proj_w, NH_ * HDIM_, 0, h1, HID_, 0, hidden,
      NH_ * HDIM_);

  // 5. LN2
  ln_kernel<<<T_, 256, 0, stream>>>(h1, ln2_w, ln2_b, ln2_bf);

  // 6. Gate logits
  gate_kernel<<<T_, 64, 0, stream>>>(ln2_bf, gate_w, logits);

  // 7. Top-2 routing with capacity
  route_kernel<<<1, 256, 0, stream>>>(logits, idx1, idx2, g1, g2, pos1, pos2,
                                      slot_tok);

  // 8. Dispatch tokens into expert slots
  dispatch_kernel<<<E_ * CAP_, 256, 0, stream>>>(ln2_bf, slot_tok, xe_bf);

  // 9. Expert FFN 1: he = gelu(xe @ w1[e]), B layout [K,N]
  gemm_bf16_kernel<1, 2><<<dim3(FFN_ / 128, CAP_ / 128, E_), 256, 0, stream>>>(
      xe_bf, HID_, (long)CAP_ * HID_, w1, FFN_, (long)HID_ * FFN_, he_bf, FFN_,
      (long)CAP_ * FFN_, nullptr, HID_);

  // 10. Expert FFN 2: eo = he @ w2[e], B layout [K,N], fp32 out
  gemm_bf16_kernel<1, 3><<<dim3(HID_ / 128, CAP_ / 128, E_), 256, 0, stream>>>(
      he_bf, FFN_, (long)CAP_ * FFN_, w2, HID_, (long)FFN_ * HID_, eo, HID_,
      (long)CAP_ * HID_, nullptr, FFN_);

  // 11. Combine: out = h1 + gated expert outputs
  combine_kernel<<<T_, 256, 0, stream>>>(h1, eo, idx1, idx2, pos1, pos2, g1, g2,
                                         out);
}
